// PGJANET_21388937134498
// MI455X (gfx1250) — matrix operand order repaired and checked by reading, not compiled, not run on hardware
//
#include <hip/hip_runtime.h>

typedef __attribute__((ext_vector_type(16))) _Float16 v16h;
typedef __attribute__((ext_vector_type(8)))  _Float16 v8h;
typedef __attribute__((ext_vector_type(8)))  float    v8f;

#define BATCH  256
#define SEQ    2048
#define HDIM   64
#define TILE_M 16
#define NWAVES 4
#define LOG2E  1.44269504088896340736f
#define XPF    16   // x prefetch distance (time steps)

__device__ __forceinline__ v8f wmma16(v16h a, v16h b, v8f c) {
  return __builtin_amdgcn_wmma_f32_16x16x32_f16(false, a, false, b, (short)0, c, false, false);
}

#if __has_builtin(__builtin_amdgcn_tanhf)
__device__ __forceinline__ float ftanh(float xv) {
  return __builtin_amdgcn_tanhf(xv);     // V_TANH_F32 (gfx1250 trans op)
}
#else
__device__ __forceinline__ float ftanh(float xv) {
  // tanh(x) = 1 - 2/(1 + e^{2x}); saturates correctly, branch-free
  return 1.0f - 2.0f * __builtin_amdgcn_rcpf(1.0f + __builtin_amdgcn_exp2f(2.0f * LOG2E * xv));
}
#endif
__device__ __forceinline__ float fsigmoid(float xv) {
  return fmaf(0.5f, ftanh(0.5f * xv), 0.5f);
}

// A fragment (16x32 f16): lane row m = lane&15; K pattern (contiguous, matches B):
// j=0..15 -> k0+j ; k0 = 16*(lane>>4) + 32*kf
__device__ __forceinline__ v16h load_afrag(const _Float16* buf, int lane, int kf) {
  int m  = lane & 15;
  int k0 = ((lane >> 4) << 4) + (kf << 5);
  const _Float16* p = buf + m * HDIM + k0;
  v8h lo = *(const v8h*)(p);
  v8h hv = *(const v8h*)(p + 8);
  return __builtin_shufflevector(lo, hv, 0,1,2,3,4,5,6,7,8,9,10,11,12,13,14,15);
}

// B fragment (32x16 f16) built once from global row-major weights W[k][n]:
// lane holds col n = lane&15 (+16*nt), k = j + 16*(lane>>4) + 32*kf
__device__ __forceinline__ v16h make_bfrag(const float* __restrict__ Wsrc,
                                           int lane, int nt, int kf) {
  int n  = (lane & 15) + (nt << 4);
  int k0 = ((lane >> 4) << 4) + (kf << 5);
  v16h f;
#pragma unroll
  for (int j = 0; j < 16; ++j)
    f[j] = (_Float16)Wsrc[(k0 + j) * HDIM + n];
  return f;
}

__global__ __launch_bounds__(32 * NWAVES) void pgjanet_kernel(
    const float* __restrict__ x,    const float* __restrict__ h0,
    const float* __restrict__ Wa,   const float* __restrict__ ba,
    const float* __restrict__ Wp1,  const float* __restrict__ bp1,
    const float* __restrict__ Wp2,  const float* __restrict__ bp2,
    const float* __restrict__ Wz,   const float* __restrict__ bz,
    const float* __restrict__ Wh,   const float* __restrict__ bh,
    const float* __restrict__ Wout, const float* __restrict__ b_out,
    float* __restrict__ out)
{
  __shared__ __align__(32) _Float16 hbuf[TILE_M * HDIM];
  __shared__ __align__(32) _Float16 ubuf[TILE_M * HDIM];
  __shared__ float4 sbuf[TILE_M];          // {amp, cos, sin, 0} per row

  const int tid  = threadIdx.x;
  const int w    = tid >> 5;               // wave id == N-tile index nt (0..3)
  const int lane = tid & 31;
  const int hi   = lane >> 4;              // half-wave (row group)
  const int n0   = lane & 15;              // column within 16-wide tile
  const int n    = n0 + (w << 4);          // this wave's global column 0..63
  const int b0   = blockIdx.x * TILE_M;

  // ---- loop-invariant weight B-fragments, register-resident (f16) ----
  const v16h fa0  = make_bfrag(Wa  + HDIM,        lane, w, 0);
  const v16h fa1  = make_bfrag(Wa  + HDIM,        lane, w, 1);
  const v16h fp10 = make_bfrag(Wp1 + HDIM,        lane, w, 0);
  const v16h fp11 = make_bfrag(Wp1 + HDIM,        lane, w, 1);
  const v16h fp20 = make_bfrag(Wp2 + HDIM,        lane, w, 0);
  const v16h fp21 = make_bfrag(Wp2 + HDIM,        lane, w, 1);
  const v16h fzu0 = make_bfrag(Wz,                lane, w, 0);
  const v16h fzu1 = make_bfrag(Wz,                lane, w, 1);
  const v16h fzh0 = make_bfrag(Wz + HDIM * HDIM,  lane, w, 0);
  const v16h fzh1 = make_bfrag(Wz + HDIM * HDIM,  lane, w, 1);
  const v16h fhu0 = make_bfrag(Wh,                lane, w, 0);
  const v16h fhu1 = make_bfrag(Wh,                lane, w, 1);
  const v16h fhh0 = make_bfrag(Wh + HDIM * HDIM,  lane, w, 0);
  const v16h fhh1 = make_bfrag(Wh + HDIM * HDIM,  lane, w, 1);

  // ---- per-lane loop-invariant scalars (this wave's column n) ----
  const float wa0  = Wa[n],  wp10s = Wp1[n], wp20s = Wp2[n];
  const float ba_r = ba[n],  bp1_r = bp1[n], bp2_r = bp2[n];
  const float bz_r = bz[n],  bh_r = bh[n];

  // ---- W_out as register B fragments (only wave 0 uses them) ----
  v16h boutf[2];
#pragma unroll
  for (int kf = 0; kf < 2; ++kf) {
    int k0 = ((lane >> 4) << 4) + (kf << 5);
    v16h f;
#pragma unroll
    for (int j = 0; j < 16; ++j) {
      float wv = (n0 < 2) ? Wout[(k0 + j) * 2 + n0] : 0.0f;
      f[j] = (_Float16)wv;
    }
    boutf[kf] = f;
  }
  const float bo_s = (n0 < 2) ? b_out[n0] : 0.0f;

  // ---- initial hidden state slice (16 rows x this wave's 16 cols) ----
  v8f h_c;
#pragma unroll
  for (int r = 0; r < 8; ++r)
    h_c[r] = h0[(size_t)(b0 + r + 8 * hi) * HDIM + n];

  // ================= recurrent scan =================
#pragma unroll 1
  for (int t = 0; t < SEQ; ++t) {
    // stage this wave's h slice (f16); wave-0 lanes stage input scalars
#pragma unroll
    for (int r = 0; r < 8; ++r)
      hbuf[(r + 8 * hi) * HDIM + n] = (_Float16)h_c[r];
    if (tid < TILE_M) {
      const float* xp = x + ((size_t)(b0 + tid) * SEQ + t) * 2;
      float ix = xp[0], qx = xp[1];
      float amp = sqrtf(fmaf(ix, ix, qx * qx));
      float inv = (amp > 0.0f) ? __builtin_amdgcn_rcpf(amp) : 0.0f;
      float cv  = (amp > 0.0f) ? ix * inv : 1.0f;
      float sv_ = qx * inv;
      sbuf[tid] = make_float4(amp, cv, sv_, 0.0f);
      if (t + XPF < SEQ) __builtin_prefetch(xp + 2 * XPF, 0, 0);
    }
    __syncthreads();                       // barrier 1: h + scalars visible

    v16h ah0 = load_afrag(hbuf, lane, 0);
    v16h ah1 = load_afrag(hbuf, lane, 1);

    // out[t-1] = hs[t-1] @ W_out + b_out  (wave 0 only; h_c is hs[t-1])
    if (w == 0 && t > 0) {
      v8f oacc = {};
      oacc = wmma16(ah0, boutf[0], oacc);
      oacc = wmma16(ah1, boutf[1], oacc);
      if (n0 < 2) {
#pragma unroll
        for (int r = 0; r < 8; ++r)
          out[((size_t)(b0 + r + 8 * hi) * SEQ + (t - 1)) * 2 + n0] = oacc[r] + bo_s;
      }
    }

    float4 sv[8];
#pragma unroll
    for (int r = 0; r < 8; ++r) sv[r] = sbuf[8 * hi + r];

    // ---- gates for this wave's 16 columns -> u = prod g*(1-g) ----
    // bias + input-scalar terms are folded into the WMMA C operand.
    v8f u;
    {
      v8f acc;
#pragma unroll
      for (int r = 0; r < 8; ++r) acc[r] = fmaf(sv[r].x, wa0, ba_r);
      acc = wmma16(ah0, fa0, acc);
      acc = wmma16(ah1, fa1, acc);
#pragma unroll
      for (int r = 0; r < 8; ++r) {
        float g = ftanh(acc[r]);
        u[r] = fmaf(-g, g, g);             // g*(1-g)
      }
    }
    {
      v8f acc;
#pragma unroll
      for (int r = 0; r < 8; ++r) acc[r] = fmaf(sv[r].y, wp10s, bp1_r);
      acc = wmma16(ah0, fp10, acc);
      acc = wmma16(ah1, fp11, acc);
#pragma unroll
      for (int r = 0; r < 8; ++r) {
        float g = ftanh(acc[r]);
        u[r] *= fmaf(-g, g, g);
      }
    }
    {
      v8f acc;
#pragma unroll
      for (int r = 0; r < 8; ++r) acc[r] = fmaf(sv[r].z, wp20s, bp2_r);
      acc = wmma16(ah0, fp20, acc);
      acc = wmma16(ah1, fp21, acc);
#pragma unroll
      for (int r = 0; r < 8; ++r) {
        float g = ftanh(acc[r]);
        u[r] *= fmaf(-g, g, g);
      }
    }

    // stage u slice (f16)
#pragma unroll
    for (int r = 0; r < 8; ++r)
      ubuf[(r + 8 * hi) * HDIM + n] = (_Float16)u[r];
    __syncthreads();                       // barrier 2: u visible

    v16h au0 = load_afrag(ubuf, lane, 0);
    v16h au1 = load_afrag(ubuf, lane, 1);

    // ---- z / h_cand / blend for this wave's 16 columns ----
    v8f accz, acch;
#pragma unroll
    for (int r = 0; r < 8; ++r) { accz[r] = bz_r; acch[r] = bh_r; }
    accz = wmma16(au0, fzu0, accz);
    accz = wmma16(au1, fzu1, accz);
    accz = wmma16(ah0, fzh0, accz);
    accz = wmma16(ah1, fzh1, accz);
    acch = wmma16(au0, fhu0, acch);
    acch = wmma16(au1, fhu1, acch);
    acch = wmma16(ah0, fhh0, acch);
    acch = wmma16(ah1, fhh1, acch);
#pragma unroll
    for (int r = 0; r < 8; ++r) {
      float zz = fsigmoid(accz[r]);
      float hc = ftanh(acch[r]);
      h_c[r] = fmaf(zz, h_c[r] - hc, hc);  // z*h + (1-z)*hc
    }
    // no trailing barrier needed: next iteration's barrier 1 orders the ubuf
    // WAR, and barrier 2 above already ordered the hbuf WAR.
  }

  // ---- epilogue: out[S-1] from final h ----
#pragma unroll
  for (int r = 0; r < 8; ++r)
    hbuf[(r + 8 * hi) * HDIM + n] = (_Float16)h_c[r];
  __syncthreads();
  if (w == 0) {
    v16h ah0 = load_afrag(hbuf, lane, 0);
    v16h ah1 = load_afrag(hbuf, lane, 1);
    v8f oacc = {};
    oacc = wmma16(ah0, boutf[0], oacc);
    oacc = wmma16(ah1, boutf[1], oacc);
    if (n0 < 2) {
#pragma unroll
      for (int r = 0; r < 8; ++r)
        out[((size_t)(b0 + r + 8 * hi) * SEQ + (SEQ - 1)) * 2 + n0] = oacc[r] + bo_s;
    }
  }
}

extern "C" void kernel_launch(void* const* d_in, const int* in_sizes, int n_in,
                              void* d_out, int out_size, void* d_ws, size_t ws_size,
                              hipStream_t stream) {
  (void)in_sizes; (void)n_in; (void)out_size; (void)d_ws; (void)ws_size;
  const float* x    = (const float*)d_in[0];
  const float* h0   = (const float*)d_in[1];
  const float* Wa   = (const float*)d_in[2];
  const float* ba   = (const float*)d_in[3];
  const float* Wp1  = (const float*)d_in[4];
  const float* bp1  = (const float*)d_in[5];
  const float* Wp2  = (const float*)d_in[6];
  const float* bp2  = (const float*)d_in[7];
  const float* Wz   = (const float*)d_in[8];
  const float* bz   = (const float*)d_in[9];
  const float* Wh   = (const float*)d_in[10];
  const float* bh   = (const float*)d_in[11];
  const float* Wout = (const float*)d_in[12];
  const float* bo   = (const float*)d_in[13];
  float* out = (float*)d_out;

  dim3 grid(BATCH / TILE_M);     // 16 batch tiles, one workgroup each
  dim3 block(32 * NWAVES);       // 4 wave32s: one per 16-column N-tile
  pgjanet_kernel<<<grid, block, 0, stream>>>(x, h0, Wa, ba, Wp1, bp1, Wp2, bp2,
                                             Wz, bz, Wh, bh, Wout, bo, out);
}
